// EnsKFstep_module_824633721259
// MI455X (gfx1250) — matrix operand order repaired and checked by reading, not compiled, not run on hardware
//
#include <hip/hip_runtime.h>

// Problem constants (from reference setup_inputs)
#define B_   8
#define ENS_ 256
#define XD_  8192
#define YD_  256

typedef __bf16 bf16_t;
typedef __attribute__((ext_vector_type(16))) __bf16 v16bf;
typedef __attribute__((ext_vector_type(8)))  __bf16 v8bf;
typedef __attribute__((ext_vector_type(2)))  float  v2f;
typedef __attribute__((ext_vector_type(8)))  float  v8f;

// ---------------------------------------------------------------------------
// WMMA wrappers (CDNA5 / gfx1250, wave32)
// ---------------------------------------------------------------------------
__device__ __forceinline__ v8f wmma_bf16(v16bf a, v16bf b, v8f c) {
  // D(f32 16x16) = A(16x32 bf16) * B(32x16 bf16) + C
  return __builtin_amdgcn_wmma_f32_16x16x32_bf16(false, a, false, b,
                                                 (short)0, c, false, false);
}
__device__ __forceinline__ v8f wmma_f32(v2f a, v2f b, v8f c) {
  // D(f32 16x16) = A(16x4 f32) * B(4x16 f32) + C
  return __builtin_amdgcn_wmma_f32_16x16x4_f32(false, a, false, b,
                                               (short)0, c, false, false);
}

// ---------------------------------------------------------------------------
// Prep: bf16 packs.  EnsBF[b][e][x], EnsTBF[b][x][e], HtBF[y][x]
// LDS-tiled 32x32 transposes so all global traffic stays coalesced.
// ---------------------------------------------------------------------------
__global__ __launch_bounds__(1024)
void k_prep_ens(const float* __restrict__ ens,
                bf16_t* __restrict__ ensbf,
                bf16_t* __restrict__ enstbf) {
  __shared__ float sh[32][33];
  int b  = blockIdx.x;
  int e0 = blockIdx.y * 32;
  int x0 = blockIdx.z * 32;
  int tx = threadIdx.x, ty = threadIdx.y;
  size_t src = ((size_t)b * ENS_ + (e0 + ty)) * XD_ + x0 + tx;
  float v = ens[src];
  ensbf[src] = (bf16_t)v;
  sh[ty][tx] = v;
  __syncthreads();
  float w = sh[tx][ty];  // = Ens[b][e0+tx][x0+ty]
  enstbf[((size_t)b * XD_ + (x0 + ty)) * ENS_ + e0 + tx] = (bf16_t)w;
}

__global__ __launch_bounds__(1024)
void k_prep_H(const float* __restrict__ H, bf16_t* __restrict__ htbf) {
  __shared__ float sh[32][33];
  int x0 = blockIdx.x * 32;
  int y0 = blockIdx.y * 32;
  int tx = threadIdx.x, ty = threadIdx.y;
  sh[ty][tx] = H[(size_t)(x0 + ty) * YD_ + y0 + tx];
  __syncthreads();
  // Ht[y][x] = H[x][y]
  htbf[(size_t)(y0 + ty) * XD_ + x0 + tx] = (bf16_t)sh[tx][ty];
}

// ---------------------------------------------------------------------------
// GEMM1: Y[b][e][y] = sum_x Ens[b][e][x] * H[x][y]   (bf16 WMMA, f32 accum)
// One wave per 32x32 tile of the 256x256 per-batch output, K = 8192.
// ---------------------------------------------------------------------------
__global__ __launch_bounds__(32)
void k_gemm_Y(const bf16_t* __restrict__ ensbf,
              const bf16_t* __restrict__ htbf,
              float* __restrict__ Ymat) {
  int b  = blockIdx.x;
  int m0 = blockIdx.y * 32;   // e
  int n0 = blockIdx.z * 32;   // y
  int lane = threadIdx.x & 31;
  int lr = lane & 15, half = lane >> 4;
  const bf16_t* A = ensbf + (size_t)b * ENS_ * XD_;   // [e][x] row-major
  v8f acc[2][2];
#pragma unroll
  for (int i = 0; i < 2; ++i)
#pragma unroll
    for (int j = 0; j < 2; ++j) acc[i][j] = (v8f)0.0f;

  for (int k0 = 0; k0 < XD_; k0 += 32) {
    v16bf af[2], bg[2];
#pragma unroll
    for (int mf = 0; mf < 2; ++mf) {
      const bf16_t* p = A + (size_t)(m0 + mf * 16 + lr) * XD_ + k0 + half * 16;
      v8bf lo = *(const v8bf*)p;
      v8bf hi = *(const v8bf*)(p + 8);
      v16bf t;
#pragma unroll
      for (int i = 0; i < 8; ++i) { t[i] = lo[i]; t[i + 8] = hi[i]; }
      af[mf] = t;
    }
#pragma unroll
    for (int nf = 0; nf < 2; ++nf) {
      const bf16_t* p = htbf + (size_t)(n0 + nf * 16 + lr) * XD_ + k0 + half * 16;
      bg[nf] = *(const v16bf*)p;   // [N][K]-major -> contiguous K
    }
#pragma unroll
    for (int mf = 0; mf < 2; ++mf)
#pragma unroll
      for (int nf = 0; nf < 2; ++nf)
        acc[mf][nf] = wmma_bf16(af[mf], bg[nf], acc[mf][nf]);
  }
#pragma unroll
  for (int mf = 0; mf < 2; ++mf)
#pragma unroll
    for (int nf = 0; nf < 2; ++nf)
#pragma unroll
      for (int r = 0; r < 8; ++r) {
        int row = m0 + mf * 16 + half * 8 + r;
        int col = n0 + nf * 16 + lr;
        Ymat[((size_t)b * ENS_ + row) * YD_ + col] = acc[mf][nf][r];
      }
}

// ---------------------------------------------------------------------------
// Stats: y mean over ensemble, Yc (transposed [b][y][e]) and innovation.
// ---------------------------------------------------------------------------
__global__ __launch_bounds__(256)
void k_stats(const float* __restrict__ Ymat,
             const float* __restrict__ ymean_t,
             const float* __restrict__ ystd_t,
             const float* __restrict__ noise,
             float* __restrict__ YcT,
             float* __restrict__ Innov) {
  __shared__ float red[256];
  int b = blockIdx.x / YD_;
  int y = blockIdx.x % YD_;
  int e = threadIdx.x;
  float v = Ymat[((size_t)b * ENS_ + e) * YD_ + y];
  red[e] = v;
  __syncthreads();
  for (int s = 128; s > 0; s >>= 1) {
    if (e < s) red[e] += red[e + s];
    __syncthreads();
  }
  float mean = red[0] * (1.0f / ENS_);
  float yc = v - mean;
  size_t idx = ((size_t)b * YD_ + y) * ENS_ + e;
  YcT[idx] = yc;
  float sd = ystd_t[y];
  Innov[idx] = ymean_t[y] - yc + noise[idx] * sd * sd;
}

// ---------------------------------------------------------------------------
// A = diag(std^2) + Yc*Yc^T / ENS      (f32 WMMA 16x16x4, full precision)
// ---------------------------------------------------------------------------
__global__ __launch_bounds__(32)
void k_cyy(const float* __restrict__ YcT, const float* __restrict__ ystd,
           float* __restrict__ Amat) {
  int b = blockIdx.x, m0 = blockIdx.y * 16, n0 = blockIdx.z * 16;
  int lane = threadIdx.x & 31;
  int lr = lane & 15, half = lane >> 4;
  const float* Yb = YcT + (size_t)b * YD_ * ENS_;
  v8f acc = (v8f)0.0f;
  for (int k = 0; k < ENS_; k += 4) {
    int kk = k + half * 2;
    v2f a, bb;
    a[0]  = Yb[(size_t)(m0 + lr) * ENS_ + kk];
    a[1]  = Yb[(size_t)(m0 + lr) * ENS_ + kk + 1];
    bb[0] = Yb[(size_t)(n0 + lr) * ENS_ + kk];
    bb[1] = Yb[(size_t)(n0 + lr) * ENS_ + kk + 1];
    acc = wmma_f32(a, bb, acc);
  }
#pragma unroll
  for (int r = 0; r < 8; ++r) {
    int row = m0 + half * 8 + r, col = n0 + lr;
    float d = (row == col) ? ystd[row] * ystd[row] : 0.0f;
    Amat[((size_t)b * YD_ + row) * YD_ + col] = acc[r] * (1.0f / ENS_) + d;
  }
}

// ---------------------------------------------------------------------------
// In-place Gauss-Jordan inverse of A (SPD, no pivoting needed). 1 block/batch.
// ---------------------------------------------------------------------------
__global__ __launch_bounds__(1024)
void k_invert(float* __restrict__ Amat) {
  __shared__ float pivrow[YD_];
  __shared__ float pivval;
  float* M = Amat + (size_t)blockIdx.x * YD_ * YD_;
  int t = threadIdx.x;
  int row = t >> 2;          // 0..255
  int c0  = (t & 3) * 64;    // 64-column slice
  for (int k = 0; k < YD_; ++k) {
    if (t == 0) pivval = 1.0f / M[k * YD_ + k];
    __syncthreads();
    float piv = pivval;
    if (row == k) {
      for (int j = c0; j < c0 + 64; ++j) pivrow[j] = M[k * YD_ + j] * piv;
    }
    __syncthreads();
    if (row == k) {
      for (int j = c0; j < c0 + 64; ++j)
        M[k * YD_ + j] = (j == k) ? piv : pivrow[j];
    } else {
      float f = M[row * YD_ + k];
      for (int j = c0; j < c0 + 64; ++j)
        if (j != k) M[row * YD_ + j] -= f * pivrow[j];
      if (k >= c0 && k < c0 + 64) M[row * YD_ + k] = -f * piv;
    }
    __syncthreads();
  }
}

// ---------------------------------------------------------------------------
// Mtmp[b][y][e] = sum_y' Ainv[y][y'] * Innov[y'][e]   (f32 WMMA)
// ---------------------------------------------------------------------------
__global__ __launch_bounds__(32)
void k_mtmp(const float* __restrict__ Ainv, const float* __restrict__ Innov,
            float* __restrict__ Mtmp) {
  int b = blockIdx.x, m0 = blockIdx.y * 16, n0 = blockIdx.z * 16;
  int lane = threadIdx.x & 31;
  int lr = lane & 15, half = lane >> 4;
  const float* Ab = Ainv + (size_t)b * YD_ * YD_;
  const float* Ib = Innov + (size_t)b * YD_ * ENS_;
  v8f acc = (v8f)0.0f;
  for (int k = 0; k < YD_; k += 4) {
    int kk = k + half * 2;
    v2f a, bb;
    a[0]  = Ab[(size_t)(m0 + lr) * YD_ + kk];
    a[1]  = Ab[(size_t)(m0 + lr) * YD_ + kk + 1];
    bb[0] = Ib[(size_t)kk * ENS_ + n0 + lr];
    bb[1] = Ib[(size_t)(kk + 1) * ENS_ + n0 + lr];
    acc = wmma_f32(a, bb, acc);
  }
#pragma unroll
  for (int r = 0; r < 8; ++r) {
    int row = m0 + half * 8 + r, col = n0 + lr;
    Mtmp[((size_t)b * YD_ + row) * ENS_ + col] = acc[r];
  }
}

// ---------------------------------------------------------------------------
// Gt[b][e][e'] = (1/ENS) * sum_y Mtmp[y][e] * Yc[y][e']   -> bf16
// ---------------------------------------------------------------------------
__global__ __launch_bounds__(32)
void k_gt(const float* __restrict__ Mtmp, const float* __restrict__ YcT,
          bf16_t* __restrict__ gtbf) {
  int b = blockIdx.x, m0 = blockIdx.y * 16, n0 = blockIdx.z * 16;
  int lane = threadIdx.x & 31;
  int lr = lane & 15, half = lane >> 4;
  const float* Mb = Mtmp + (size_t)b * YD_ * ENS_;
  const float* Yb = YcT + (size_t)b * YD_ * ENS_;
  v8f acc = (v8f)0.0f;
  for (int k = 0; k < YD_; k += 4) {
    int kk = k + half * 2;
    v2f a, bb;
    a[0]  = Mb[(size_t)kk * ENS_ + m0 + lr];
    a[1]  = Mb[(size_t)(kk + 1) * ENS_ + m0 + lr];
    bb[0] = Yb[(size_t)kk * ENS_ + n0 + lr];
    bb[1] = Yb[(size_t)(kk + 1) * ENS_ + n0 + lr];
    acc = wmma_f32(a, bb, acc);
  }
#pragma unroll
  for (int r = 0; r < 8; ++r) {
    int row = m0 + half * 8 + r, col = n0 + lr;
    gtbf[((size_t)b * ENS_ + row) * ENS_ + col] = (bf16_t)(acc[r] * (1.0f / ENS_));
  }
}

// ---------------------------------------------------------------------------
// Final: out[b][e][x] = Ens[b][e][x] + sum_e' Gt[e][e'] * Ens[b][e'][x]
// (bf16 WMMA on the update term, f32 state add). B operand from EnsTBF.
// ---------------------------------------------------------------------------
__global__ __launch_bounds__(32)
void k_final(const float* __restrict__ ens,
             const bf16_t* __restrict__ gtbf,
             const bf16_t* __restrict__ enstbf,
             float* __restrict__ out) {
  int b  = blockIdx.x;
  int m0 = blockIdx.y * 32;   // e
  int n0 = blockIdx.z * 32;   // x
  int lane = threadIdx.x & 31;
  int lr = lane & 15, half = lane >> 4;
  const bf16_t* A  = gtbf + (size_t)b * ENS_ * ENS_;     // [e][e']
  const bf16_t* Bt = enstbf + (size_t)b * XD_ * ENS_;    // [x][e']
  v8f acc[2][2];
#pragma unroll
  for (int i = 0; i < 2; ++i)
#pragma unroll
    for (int j = 0; j < 2; ++j) acc[i][j] = (v8f)0.0f;

  for (int k0 = 0; k0 < ENS_; k0 += 32) {
    v16bf af[2], bg[2];
#pragma unroll
    for (int mf = 0; mf < 2; ++mf) {
      const bf16_t* p = A + (size_t)(m0 + mf * 16 + lr) * ENS_ + k0 + half * 16;
      v8bf lo = *(const v8bf*)p;
      v8bf hi = *(const v8bf*)(p + 8);
      v16bf t;
#pragma unroll
      for (int i = 0; i < 8; ++i) { t[i] = lo[i]; t[i + 8] = hi[i]; }
      af[mf] = t;
    }
#pragma unroll
    for (int nf = 0; nf < 2; ++nf) {
      const bf16_t* p = Bt + (size_t)(n0 + nf * 16 + lr) * ENS_ + k0 + half * 16;
      bg[nf] = *(const v16bf*)p;
    }
#pragma unroll
    for (int mf = 0; mf < 2; ++mf)
#pragma unroll
      for (int nf = 0; nf < 2; ++nf)
        acc[mf][nf] = wmma_bf16(af[mf], bg[nf], acc[mf][nf]);
  }
#pragma unroll
  for (int mf = 0; mf < 2; ++mf)
#pragma unroll
    for (int nf = 0; nf < 2; ++nf)
#pragma unroll
      for (int r = 0; r < 8; ++r) {
        int row = m0 + mf * 16 + half * 8 + r;   // e
        int col = n0 + nf * 16 + lr;             // x
        size_t gi = ((size_t)b * ENS_ + row) * XD_ + col;
        out[gi] = ens[gi] + acc[mf][nf][r];
      }
}

// ---------------------------------------------------------------------------
// Host launcher
// ---------------------------------------------------------------------------
extern "C" void kernel_launch(void* const* d_in, const int* in_sizes, int n_in,
                              void* d_out, int out_size, void* d_ws, size_t ws_size,
                              hipStream_t stream) {
  const float* Ens   = (const float*)d_in[0];  // [8,256,8192]
  const float* H     = (const float*)d_in[1];  // [8192,256]
  const float* ymean = (const float*)d_in[2];  // [1,256]
  const float* ystd  = (const float*)d_in[3];  // [1,256]
  const float* noise = (const float*)d_in[4];  // [8,256,256]
  float* out = (float*)d_out;

  // Workspace carve-up (all 256B-aligned; total ~79 MB)
  char* ws = (char*)d_ws;
  bf16_t* ensbf  = (bf16_t*)(ws + 0);           // 33,554,432
  bf16_t* enstbf = (bf16_t*)(ws + 33554432);    // 33,554,432
  bf16_t* htbf   = (bf16_t*)(ws + 67108864);    //  4,194,304
  float*  Ymat   = (float*) (ws + 71303168);    //  2,097,152
  float*  YcT    = (float*) (ws + 73400320);    //  2,097,152
  float*  Innov  = (float*) (ws + 75497472);    //  2,097,152
  float*  Amat   = (float*) (ws + 77594624);    //  2,097,152
  float*  Mtmp   = (float*) (ws + 79691776);    //  2,097,152
  bf16_t* gtbf   = (bf16_t*)(ws + 81788928);    //  1,048,576

  k_prep_ens<<<dim3(B_, ENS_ / 32, XD_ / 32), dim3(32, 32), 0, stream>>>(Ens, ensbf, enstbf);
  k_prep_H<<<dim3(XD_ / 32, YD_ / 32), dim3(32, 32), 0, stream>>>(H, htbf);
  k_gemm_Y<<<dim3(B_, ENS_ / 32, YD_ / 32), 32, 0, stream>>>(ensbf, htbf, Ymat);
  k_stats<<<B_ * YD_, ENS_, 0, stream>>>(Ymat, ymean, ystd, noise, YcT, Innov);
  k_cyy<<<dim3(B_, YD_ / 16, YD_ / 16), 32, 0, stream>>>(YcT, ystd, Amat);
  k_invert<<<B_, 1024, 0, stream>>>(Amat);
  k_mtmp<<<dim3(B_, YD_ / 16, ENS_ / 16), 32, 0, stream>>>(Amat, Innov, Mtmp);
  k_gt<<<dim3(B_, ENS_ / 16, ENS_ / 16), 32, 0, stream>>>(Mtmp, YcT, gtbf);
  k_final<<<dim3(B_, ENS_ / 32, XD_ / 32), 32, 0, stream>>>(Ens, gtbf, enstbf, out);
}
